// GCN_2302102470991
// MI455X (gfx1250) — matrix operand order repaired and checked by reading, not compiled, not run on hardware
//
#include <hip/hip_runtime.h>
#include <hip/hip_bf16.h>

#define N_NODES 50000
#define N_EDGES 600000
#define IN_DIM  128
#define OUT_DIM 10
#define M_TILES (N_NODES / 16)   // 3125, exact

typedef __attribute__((ext_vector_type(16))) __bf16 v16bf;
typedef __attribute__((ext_vector_type(8)))  float  v8f;

// ---------------------------------------------------------------------------
// Prep: deg = 1.0 (self loop), pooled = 0, split W_gcn into bf16 hi/lo planes.
// ---------------------------------------------------------------------------
__global__ void __launch_bounds__(256) gcn_prep(const float* __restrict__ Wg,
                                                float* __restrict__ deg,
                                                float* __restrict__ pooled,
                                                __bf16* __restrict__ Whi,
                                                __bf16* __restrict__ Wlo) {
    int idx = blockIdx.x * 256 + threadIdx.x;
    if (idx < N_NODES) deg[idx] = 1.0f;
    if (idx < IN_DIM)  pooled[idx] = 0.0f;
    if (idx < IN_DIM * IN_DIM) {
        float w = Wg[idx];
        __bf16 hi = (__bf16)w;
        Whi[idx] = hi;
        Wlo[idx] = (__bf16)(w - (float)hi);
    }
}

// ---------------------------------------------------------------------------
// Degree accumulation over edges (aggregation index = ei[0]).
// ---------------------------------------------------------------------------
__global__ void __launch_bounds__(256) gcn_deg(const int* __restrict__ ei0,
                                               float* __restrict__ deg) {
    int e = blockIdx.x * 256 + threadIdx.x;
    if (e < N_EDGES) atomicAdd(&deg[ei0[e]], 1.0f);
}

__global__ void __launch_bounds__(256) gcn_dis(const float* __restrict__ deg,
                                               float* __restrict__ dis) {
    int n = blockIdx.x * 256 + threadIdx.x;
    if (n < N_NODES) dis[n] = rsqrtf(deg[n]);   // deg >= 1 always
}

// ---------------------------------------------------------------------------
// h = x @ W_gcn^T  via v_wmma_f32_16x16x32_bf16, bf16x3 split for ~fp32 acc.
// One block = 8 waves; wave w owns output n-tile w; block owns one m-tile.
// A (x tile, fp32) staged in LDS with padded stride; B read from prepacked
// bf16 hi/lo W planes (B[k][n] = W[n][k]).
// ---------------------------------------------------------------------------
__global__ void __launch_bounds__(256) gcn_gemm(const float* __restrict__ x,
                                                const __bf16* __restrict__ Whi,
                                                const __bf16* __restrict__ Wlo,
                                                float* __restrict__ h) {
    __shared__ float xs[16 * 132];          // 16 rows x 128 cols, stride 132
    const int mtile = blockIdx.x;
    const int t = threadIdx.x;

    // Cooperative coalesced load of the 16x128 x-tile (float4 granularity).
    const float4* xsrc = (const float4*)(x + (size_t)mtile * 16 * IN_DIM);
#pragma unroll
    for (int i = 0; i < 2; ++i) {
        int vid = t + i * 256;              // 0..511 float4s
        int row = vid >> 5;                 // 32 float4 per row
        int c4  = vid & 31;
        float4 v = xsrc[row * 32 + c4];
        float* dst = &xs[row * 132 + c4 * 4];   // 132 % 4 == 0 -> aligned
        dst[0] = v.x; dst[1] = v.y; dst[2] = v.z; dst[3] = v.w;
    }
    __syncthreads();

    const int wave  = t >> 5;               // n-tile id 0..7
    const int lane  = t & 31;
    const int half  = lane >> 4;            // K half-lane split
    const int mr    = lane & 15;            // A row within tile
    const int nglob = wave * 16 + (lane & 15);  // output column / W row

    v8f acc = {};
    for (int kc = 0; kc < 4; ++kc) {        // K chunks of 32
        v16bf ahi, alo, bhi, blo;
#pragma unroll
        for (int j = 0; j < 8; ++j) {
            // A 16-bit 16x32 layout (ISA 7.12.2)
            int kA = kc * 32 + ((j >= 4) ? 16 : 0) + half * 8 + (j & 3) * 2;
            float f0 = xs[mr * 132 + kA];
            float f1 = xs[mr * 132 + kA + 1];
            __bf16 h0 = (__bf16)f0, h1 = (__bf16)f1;
            ahi[2 * j]     = h0;
            ahi[2 * j + 1] = h1;
            alo[2 * j]     = (__bf16)(f0 - (float)h0);
            alo[2 * j + 1] = (__bf16)(f1 - (float)h1);
            // B 16-bit 32x16 layout: same K order as A
            int kB = kc * 32 + ((j >= 4) ? 16 : 0) + half * 8 + (j & 3) * 2;
            const __bf16* wh = Whi + (size_t)nglob * IN_DIM + kB;
            const __bf16* wl = Wlo + (size_t)nglob * IN_DIM + kB;
            bhi[2 * j] = wh[0]; bhi[2 * j + 1] = wh[1];
            blo[2 * j] = wl[0]; blo[2 * j + 1] = wl[1];
        }
        // bf16x3: x*w ~= a_lo*b_hi + a_hi*b_lo + a_hi*b_hi (f32 accumulate)
        acc = __builtin_amdgcn_wmma_f32_16x16x32_bf16(false, alo, false, bhi,
                                                      (short)0, acc, false, false);
        acc = __builtin_amdgcn_wmma_f32_16x16x32_bf16(false, ahi, false, blo,
                                                      (short)0, acc, false, false);
        acc = __builtin_amdgcn_wmma_f32_16x16x32_bf16(false, ahi, false, bhi,
                                                      (short)0, acc, false, false);
    }

    // C/D layout: VGPR r -> M = half*8 + r, N = lane&15
#pragma unroll
    for (int r = 0; r < 8; ++r) {
        h[((size_t)mtile * 16 + half * 8 + r) * IN_DIM + nglob] = acc[r];
    }
}

// ---------------------------------------------------------------------------
// out[n][:] = h[n][:] * (dis[n]^2)   (self-loop message; also initializes out)
// ---------------------------------------------------------------------------
__global__ void __launch_bounds__(256) gcn_self(const float* __restrict__ h,
                                                const float* __restrict__ dis,
                                                float* __restrict__ out) {
    size_t idx = (size_t)blockIdx.x * 256 + threadIdx.x;
    if (idx < (size_t)N_NODES * IN_DIM) {
        int n = (int)(idx >> 7);
        float d = dis[n];
        out[idx] = h[idx] * (d * d);
    }
}

// ---------------------------------------------------------------------------
// Edge scatter: one wave32 per edge. Each lane gathers 4 contiguous floats of
// h[j] (float4) and atomically accumulates into out[i].
// ---------------------------------------------------------------------------
__global__ void __launch_bounds__(256) gcn_edge(const int* __restrict__ ei0,
                                                const int* __restrict__ ei1,
                                                const float* __restrict__ dis,
                                                const float* __restrict__ h,
                                                float* __restrict__ out) {
    int e    = (blockIdx.x * 256 + threadIdx.x) >> 5;
    int lane = threadIdx.x & 31;
    if (e >= N_EDGES) return;
    int i = ei0[e];
    int j = ei1[e];
    float w = dis[i] * dis[j];
    const float4* hj = (const float4*)(h + (size_t)j * IN_DIM);
    float4 v = hj[lane];                    // 32 lanes x 4 floats = 128
    float* orow = out + (size_t)i * IN_DIM + lane * 4;
    atomicAdd(orow + 0, w * v.x);
    atomicAdd(orow + 1, w * v.y);
    atomicAdd(orow + 2, w * v.z);
    atomicAdd(orow + 3, w * v.w);
}

// ---------------------------------------------------------------------------
// Column partial sums: 64 strided partials per column -> 8192 atomics total.
// ---------------------------------------------------------------------------
__global__ void __launch_bounds__(256) gcn_pool(const float* __restrict__ out,
                                                float* __restrict__ pooled) {
    int tid = blockIdx.x * 256 + threadIdx.x;   // 0..8191
    if (tid >= IN_DIM * 64) return;
    int d = tid & (IN_DIM - 1);
    int s = tid >> 7;
    float acc = 0.0f;
    for (int n = s; n < N_NODES; n += 64)
        acc += out[(size_t)n * IN_DIM + d];
    atomicAdd(&pooled[d], acc);
}

// ---------------------------------------------------------------------------
// pooled/N + b_gcn -> 10x128 matvec + b_out -> log_softmax -> d_out[10]
// ---------------------------------------------------------------------------
__global__ void __launch_bounds__(128) gcn_head(const float* __restrict__ pooled,
                                                const float* __restrict__ b_gcn,
                                                const float* __restrict__ W_out,
                                                const float* __restrict__ b_out,
                                                float* __restrict__ dout) {
    __shared__ float p[IN_DIM];
    __shared__ float lg[OUT_DIM];
    int t = threadIdx.x;
    p[t] = pooled[t] * (1.0f / (float)N_NODES) + b_gcn[t];
    __syncthreads();
    if (t < OUT_DIM) {
        float acc = b_out[t];
        for (int k = 0; k < IN_DIM; ++k)
            acc += p[k] * W_out[t * IN_DIM + k];
        lg[t] = acc;
    }
    __syncthreads();
    if (t < OUT_DIM) {
        float m = lg[0];
        for (int o = 1; o < OUT_DIM; ++o) m = fmaxf(m, lg[o]);
        float s = 0.0f;
        for (int o = 0; o < OUT_DIM; ++o) s += expf(lg[o] - m);
        dout[t] = lg[t] - m - logf(s);
    }
}

// ---------------------------------------------------------------------------
extern "C" void kernel_launch(void* const* d_in, const int* in_sizes, int n_in,
                              void* d_out, int out_size, void* d_ws, size_t ws_size,
                              hipStream_t stream) {
    const float* x     = (const float*)d_in[0];
    const int*   ei    = (const int*)d_in[1];      // [2, E] flat, int32
    const float* W_gcn = (const float*)d_in[2];
    const float* b_gcn = (const float*)d_in[3];
    const float* W_out = (const float*)d_in[4];
    const float* b_out = (const float*)d_in[5];
    float*       outp  = (float*)d_out;

    const int* ei0 = ei;
    const int* ei1 = ei + N_EDGES;

    // Workspace carve-up (256 B aligned).
    char* base = (char*)d_ws;
    const size_t HO_BYTES = (size_t)N_NODES * IN_DIM * sizeof(float);   // 25,600,000
    float*  h      = (float*)(base);
    float*  aggr   = (float*)(base + HO_BYTES);
    float*  deg    = (float*)(base + 2 * HO_BYTES);
    float*  dis    = (float*)(base + 2 * HO_BYTES + 200192);
    __bf16* Whi    = (__bf16*)(base + 2 * HO_BYTES + 400384);
    __bf16* Wlo    = (__bf16*)(base + 2 * HO_BYTES + 400384 + 32768);
    float*  pooled = (float*)(base + 2 * HO_BYTES + 400384 + 65536);

    // 1. prep: deg=1, pooled=0, split W into bf16 hi/lo
    gcn_prep<<<(N_NODES + 255) / 256, 256, 0, stream>>>(W_gcn, deg, pooled, Whi, Wlo);
    // 2. degree accumulation
    gcn_deg<<<(N_EDGES + 255) / 256, 256, 0, stream>>>(ei0, deg);
    // 3. dis = rsqrt(deg)
    gcn_dis<<<(N_NODES + 255) / 256, 256, 0, stream>>>(deg, dis);
    // 4. h = x @ W^T (WMMA bf16x3)
    gcn_gemm<<<M_TILES, 256, 0, stream>>>(x, Whi, Wlo, h);
    // 5. self-loop init: out = h / deg
    gcn_self<<<(int)(((size_t)N_NODES * IN_DIM + 255) / 256), 256, 0, stream>>>(h, dis, aggr);
    // 6. edge scatter-add (1 wave per edge)
    gcn_edge<<<(N_EDGES * 32 + 255) / 256, 256, 0, stream>>>(ei0, ei1, dis, h, aggr);
    // 7. column partial sums
    gcn_pool<<<(IN_DIM * 64 + 255) / 256, 256, 0, stream>>>(aggr, pooled);
    // 8. head: mean + linear + log_softmax
    gcn_head<<<1, 128, 0, stream>>>(pooled, b_gcn, W_out, b_out, outp);
}
